// GraphRobertaCL_66194035966391
// MI455X (gfx1250) — matrix operand order repaired and checked by reading, not compiled, not run on hardware
//
#include <hip/hip_runtime.h>
#include <hip/hip_bf16.h>

// ---------------------------------------------------------------------------
// GraphRobertaCL forward for MI455X (gfx1250, wave32, WMMA).
// Heavy GEMMs + attention use v_wmma_f32_16x16x32_bf16; tile staging uses
// CDNA5 async global->LDS copies (ASYNCcnt); graph head in f32.
// ---------------------------------------------------------------------------

typedef __attribute__((ext_vector_type(16))) __bf16        v16bf;
typedef __attribute__((ext_vector_type(8)))  float          v8f;
typedef __attribute__((ext_vector_type(8)))  unsigned short ushort8;

union FragA { v16bf v; ushort8 h[2]; };

__device__ __forceinline__ unsigned short f2bf(float f) {
    unsigned int u = __float_as_uint(f);
    unsigned int r = (u + 0x7FFFu + ((u >> 16) & 1u)) >> 16;
    return (unsigned short)r;
}

// CDNA5 async global->LDS 16B copy (one per lane). lds_off = wave-relative
// LDS byte address (low 32 bits of generic shared pointer); tracked by ASYNCcnt.
__device__ __forceinline__ void async_copy_b128(unsigned lds_off, const void* gaddr) {
    asm volatile("global_load_async_to_lds_b128 %0, %1, off"
                 :: "v"(lds_off), "v"(gaddr) : "memory");
}
__device__ __forceinline__ void wait_async_lds() {
    asm volatile("s_wait_asynccnt 0x0" ::: "memory");
}

__device__ __forceinline__ float block_reduce_sum256(float v, float* red) {
    int tid = threadIdx.x;
    red[tid] = v; __syncthreads();
    for (int off = 128; off > 0; off >>= 1) {
        if (tid < off) red[tid] += red[tid + off];
        __syncthreads();
    }
    float r = red[0]; __syncthreads();
    return r;
}

// ---------------------------------------------------------------------------
// bf16 WMMA GEMM: C[M,N] = act(A[M,K] * B[K,N] + bias + resid)
// BM=128 BN=128 BK=32, 256 threads (8 waves), wave = 64x32 subtile (4x2 WMMA)
// act: 0=none 1=gelu(exact) 2=tanh
// ---------------------------------------------------------------------------
__global__ __launch_bounds__(256) void gemm_bf16_wmma(
    const unsigned short* __restrict__ A, int lda,
    const unsigned short* __restrict__ B, int ldb,
    const float* __restrict__ bias,
    const float* __restrict__ resid, int ldr,
    float* __restrict__ Cf, int ldcf,
    unsigned short* __restrict__ Cb, int ldcb,
    int K, int act)
{
    constexpr int LDSA = 40;  // 32 + 8 pad (shorts)
    constexpr int LDSB = 40;
    __shared__ __align__(16) unsigned short sA[128 * LDSA];
    __shared__ __align__(16) unsigned short sB[128 * LDSB];

    const int tid  = threadIdx.x;
    const int lane = tid & 31;
    const int wave = tid >> 5;
    const int waveM = (wave >> 2) * 64;
    const int waveN = (wave & 3) * 32;
    const int bm = blockIdx.x * 128;
    const int bn = blockIdx.y * 128;

    const int mA = lane & 15;
    const int kq = (lane >> 4) * 8;     // A K-half offset
    const int nB = lane & 15;
    const int kb = (lane >> 4) * 8;     // B K-half offset

    v8f acc[4][2];
#pragma unroll
    for (int mt = 0; mt < 4; ++mt)
#pragma unroll
        for (int nt = 0; nt < 2; ++nt)
#pragma unroll
            for (int r = 0; r < 8; ++r) acc[mt][nt][r] = 0.0f;

    const int rA = tid >> 1;            // A tile: 128 rows, 2 chunks of 16
    const int cA = (tid & 1) * 16;
    const unsigned ldsA0 = (unsigned)(size_t)(sA + rA * LDSA + cA);
    const unsigned ldsA1 = ldsA0 + 16u;

    for (int k0 = 0; k0 < K; k0 += 32) {
        // ---- stage A tile via async global->LDS (2 x 16B per thread) ----
        {
            const unsigned short* g = A + (size_t)(bm + rA) * lda + k0 + cA;
            async_copy_b128(ldsA0, g);
            async_copy_b128(ldsA1, g + 8);
            if (k0 + 32 < K)
                __builtin_prefetch(A + (size_t)(bm + rA) * lda + k0 + 32 + cA, 0, 3);
        }
        // ---- stage B tile transposed: sB[n][k] = B[k0+k][bn+n] ----
#pragma unroll
        for (int it = 0; it < 2; ++it) {
            int idx = tid + it * 256;          // 0..511
            int kk  = idx >> 4;                // 0..31
            int nc  = (idx & 15) * 8;          // 0..120
            ushort8 v = *(const ushort8*)(B + (size_t)(k0 + kk) * ldb + bn + nc);
#pragma unroll
            for (int j = 0; j < 8; ++j) sB[(nc + j) * LDSB + kk] = v[j];
            if (k0 + 32 < K)
                __builtin_prefetch(B + (size_t)(k0 + 32 + kk) * ldb + bn + nc, 0, 3);
        }
        wait_async_lds();
        __syncthreads();

        FragA af[4], bfr[2];
#pragma unroll
        for (int mt = 0; mt < 4; ++mt) {
            const unsigned short* p = sA + (waveM + mt * 16 + mA) * LDSA + kq;
            af[mt].h[0] = *(const ushort8*)p;
            af[mt].h[1] = *(const ushort8*)(p + 16);
        }
#pragma unroll
        for (int nt = 0; nt < 2; ++nt) {
            const unsigned short* p = sB + (waveN + nt * 16 + nB) * LDSB + kb;
            bfr[nt].h[0] = *(const ushort8*)p;
            bfr[nt].h[1] = *(const ushort8*)(p + 16);
        }
#pragma unroll
        for (int mt = 0; mt < 4; ++mt)
#pragma unroll
            for (int nt = 0; nt < 2; ++nt)
                acc[mt][nt] = __builtin_amdgcn_wmma_f32_16x16x32_bf16(
                    false, af[mt].v, false, bfr[nt].v, (short)0, acc[mt][nt], false, false);
        __syncthreads();
    }

    // ---- epilogue ----
    const int rowL = (lane >> 4) * 8;
    const int colL = lane & 15;
#pragma unroll
    for (int mt = 0; mt < 4; ++mt)
#pragma unroll
        for (int nt = 0; nt < 2; ++nt)
#pragma unroll
            for (int r = 0; r < 8; ++r) {
                int row = bm + waveM + mt * 16 + rowL + r;
                int col = bn + waveN + nt * 16 + colL;
                float v = acc[mt][nt][r];
                if (bias)  v += bias[col];
                if (resid) v += resid[(size_t)row * ldr + col];
                if (act == 1)      v = 0.5f * v * (1.0f + erff(v * 0.70710678118654752f));
                else if (act == 2) v = tanhf(v);
                if (Cf) Cf[(size_t)row * ldcf + col] = v;
                if (Cb) Cb[(size_t)row * ldcb + col] = f2bf(v);
            }
}

// ---------------------------------------------------------------------------
// Fused attention per (head, batch): S = QK^T/8 + mask; P = softmax(S); C = PV
// L=128, HD=64. 256 threads = 8 waves (wave w owns 16 query rows).
// ---------------------------------------------------------------------------
__global__ __launch_bounds__(256) void attn_kernel(
    const unsigned short* __restrict__ qkv,   // [N*L, 2304] bf16 (q|k|v)
    const int* __restrict__ amask,            // [N, L]
    unsigned short* __restrict__ ctx)         // [N*L, 768] bf16
{
    __shared__ __align__(16) unsigned short sQ[128 * 72];
    __shared__ __align__(16) unsigned short sK[128 * 72];
    __shared__ __align__(16) unsigned short sVt[64 * 136];
    __shared__ __align__(16) float          sS[128 * 132];   // later aliased as bf16 P

    const int h = blockIdx.x;   // 0..11
    const int b = blockIdx.y;   // 0..511
    const int tid  = threadIdx.x;
    const int lane = tid & 31;
    const int wave = tid >> 5;
    const size_t base = (size_t)b * 128 * 2304;

    // stage Q, K row-major via async copies; V transposed via scatter
    for (int c = tid; c < 128 * 8; c += 256) {
        int row = c >> 3, cc = (c & 7) * 8;
        const unsigned short* gq = qkv + base + (size_t)row * 2304 + h * 64 + cc;
        async_copy_b128((unsigned)(size_t)(sQ + row * 72 + cc), gq);
        async_copy_b128((unsigned)(size_t)(sK + row * 72 + cc), gq + 768);
        ushort8 v = *(const ushort8*)(gq + 1536);
#pragma unroll
        for (int j = 0; j < 8; ++j) sVt[(cc + j) * 136 + row] = v[j];
    }
    wait_async_lds();
    __syncthreads();

    const int mA = lane & 15, kq = (lane >> 4) * 8;
    const int nB = lane & 15, kb = (lane >> 4) * 8;
    const int rowL = (lane >> 4) * 8, colL = lane & 15;

    // ---- scores: 16x128 per wave, Kdim = 64 ----
    v8f sacc[8];
#pragma unroll
    for (int nt = 0; nt < 8; ++nt)
#pragma unroll
        for (int r = 0; r < 8; ++r) sacc[nt][r] = 0.0f;

#pragma unroll
    for (int kt = 0; kt < 2; ++kt) {
        FragA a;
        const unsigned short* p = sQ + (wave * 16 + mA) * 72 + kt * 32 + kq;
        a.h[0] = *(const ushort8*)p;
        a.h[1] = *(const ushort8*)(p + 16);
#pragma unroll
        for (int nt = 0; nt < 8; ++nt) {
            FragA bq;
            const unsigned short* pb = sK + (nt * 16 + nB) * 72 + kt * 32 + kb;
            bq.h[0] = *(const ushort8*)pb;
            bq.h[1] = *(const ushort8*)(pb + 16);
            sacc[nt] = __builtin_amdgcn_wmma_f32_16x16x32_bf16(
                false, a.v, false, bq.v, (short)0, sacc[nt], false, false);
        }
    }
#pragma unroll
    for (int nt = 0; nt < 8; ++nt)
#pragma unroll
        for (int r = 0; r < 8; ++r) {
            int row = wave * 16 + rowL + r;
            int col = nt * 16 + colL;
            float mk = (1.0f - (float)amask[b * 128 + col]) * -1e9f;
            sS[row * 132 + col] = sacc[nt][r] * 0.125f + mk;
        }
    __syncthreads();

    // ---- softmax rows, P written in-place as bf16 ----
    if (tid < 128) {
        float* Srow = sS + tid * 132;
        float mx = -1e30f;
        for (int c = 0; c < 128; ++c) mx = fmaxf(mx, Srow[c]);
        float sum = 0.0f;
        for (int c = 0; c < 128; ++c) { float e = expf(Srow[c] - mx); Srow[c] = e; sum += e; }
        float inv = 1.0f / sum;
        unsigned short* Prow = (unsigned short*)Srow;
        for (int c = 0; c < 128; ++c) Prow[c] = f2bf(Srow[c] * inv);  // 2c < 4c: safe
    }
    __syncthreads();

    // ---- context: 16x64 per wave, Kdim = 128 ----
    v8f cacc[4];
#pragma unroll
    for (int nt = 0; nt < 4; ++nt)
#pragma unroll
        for (int r = 0; r < 8; ++r) cacc[nt][r] = 0.0f;

    const unsigned short* P = (const unsigned short*)sS;  // row stride 264 shorts
#pragma unroll
    for (int kt = 0; kt < 4; ++kt) {
        FragA a;
        const unsigned short* p = P + (wave * 16 + mA) * 264 + kt * 32 + kq;
        a.h[0] = *(const ushort8*)p;
        a.h[1] = *(const ushort8*)(p + 16);
#pragma unroll
        for (int nt = 0; nt < 4; ++nt) {
            FragA bq;
            const unsigned short* pb = sVt + (nt * 16 + nB) * 136 + kt * 32 + kb;
            bq.h[0] = *(const ushort8*)pb;
            bq.h[1] = *(const ushort8*)(pb + 16);
            cacc[nt] = __builtin_amdgcn_wmma_f32_16x16x32_bf16(
                false, a.v, false, bq.v, (short)0, cacc[nt], false, false);
        }
    }
#pragma unroll
    for (int nt = 0; nt < 4; ++nt)
#pragma unroll
        for (int r = 0; r < 8; ++r) {
            int row = wave * 16 + rowL + r;
            int col = nt * 16 + colL;
            ctx[((size_t)b * 128 + row) * 768 + h * 64 + col] = f2bf(cacc[nt][r]);
        }
}

// ---------------------------------------------------------------------------
// Elementwise / reduction helpers
// ---------------------------------------------------------------------------
__global__ void cvt_f32_bf16(const float* __restrict__ in, unsigned short* __restrict__ out, size_t n) {
    for (size_t i = (size_t)blockIdx.x * blockDim.x + threadIdx.x; i < n; i += (size_t)gridDim.x * blockDim.x)
        out[i] = f2bf(in[i]);
}

__global__ void fill_f32(float* p, float v, size_t n) {
    for (size_t i = (size_t)blockIdx.x * blockDim.x + threadIdx.x; i < n; i += (size_t)gridDim.x * blockDim.x)
        p[i] = v;
}

__global__ __launch_bounds__(256) void embed_kernel(
    const int* __restrict__ ids, const float* __restrict__ we,
    const float* __restrict__ pe, const float* __restrict__ te,
    float* __restrict__ out)
{
    int tok = blockIdx.x;            // 0..65535
    int s = tok & 127;
    int id = ids[tok];
    for (int c = threadIdx.x; c < 768; c += 256)
        out[(size_t)tok * 768 + c] =
            we[(size_t)id * 768 + c] + pe[(size_t)(s + 2) * 768 + c] + te[c];
}

__global__ __launch_bounds__(256) void ln_kernel(
    const float* __restrict__ in, const float* __restrict__ sc, const float* __restrict__ bi,
    float* __restrict__ outf, unsigned short* __restrict__ outb, int D)
{
    __shared__ float red[256];
    const int row = blockIdx.x;
    const float* x = in + (size_t)row * D;
    float p = 0.0f;
    for (int c = threadIdx.x; c < D; c += 256) p += x[c];
    float mean = block_reduce_sum256(p, red) / (float)D;
    p = 0.0f;
    for (int c = threadIdx.x; c < D; c += 256) { float d = x[c] - mean; p += d * d; }
    float inv = rsqrtf(block_reduce_sum256(p, red) / (float)D + 1e-5f);
    for (int c = threadIdx.x; c < D; c += 256) {
        float v = (x[c] - mean) * inv * sc[c] + bi[c];
        if (outf) outf[(size_t)row * D + c] = v;
        if (outb) outb[(size_t)row * D + c] = f2bf(v);
    }
}

// ---------------------------------------------------------------------------
// Graph head (f32): naive GEMM, GAT segment softmax, aggregation
// ---------------------------------------------------------------------------
__global__ void sgemm_naive(const float* __restrict__ A, int lda,
                            const float* __restrict__ B, int ldb,
                            const float* __restrict__ bias,
                            float* __restrict__ C, int ldc,
                            int M, int N, int K, int act)  // act: 0 none, 1 relu, 2 tanh
{
    int idx = blockIdx.x * blockDim.x + threadIdx.x;
    if (idx >= M * N) return;
    int m = idx / N, n = idx % N;
    float s = bias ? bias[n] : 0.0f;
    for (int k = 0; k < K; ++k) s += A[(size_t)m * lda + k] * B[(size_t)k * ldb + n];
    if (act == 1) s = fmaxf(s, 0.0f);
    else if (act == 2) s = tanhf(s);
    C[(size_t)m * ldc + n] = s;
}

__global__ __launch_bounds__(256) void gat_scores(
    const float* __restrict__ hfeat, const float* __restrict__ asrc, const float* __restrict__ adst,
    float* __restrict__ asr, float* __restrict__ ads, int heads, int ch)
{
    __shared__ float red[256];
    const int n = blockIdx.x;
    for (int h = 0; h < heads; ++h) {
        float p = 0.0f, q = 0.0f;
        for (int c = threadIdx.x; c < ch; c += 256) {
            float hv = hfeat[(size_t)n * heads * ch + h * ch + c];
            p += hv * asrc[h * ch + c];
            q += hv * adst[h * ch + c];
        }
        float rp = block_reduce_sum256(p, red);
        float rq = block_reduce_sum256(q, red);
        if (threadIdx.x == 0) { asr[n * heads + h] = rp; ads[n * heads + h] = rq; }
        __syncthreads();
    }
}

__device__ __forceinline__ void atomicMaxF(float* addr, float v) {
    unsigned int* a = (unsigned int*)addr;
    unsigned int old = *a;
    while (__uint_as_float(old) < v) {
        unsigned int assumed = old;
        old = atomicCAS(a, assumed, __float_as_uint(v));
        if (old == assumed) break;
    }
}

__global__ void gat_edge_logits(const int* __restrict__ ei, int E,
                                const float* __restrict__ asr, const float* __restrict__ ads,
                                float* __restrict__ ev, float* __restrict__ mbuf,
                                int heads, int total)
{
    int idx = blockIdx.x * blockDim.x + threadIdx.x;
    if (idx >= total) return;
    int e = idx / heads, h = idx % heads;
    int s, d;
    if (e < E) { s = ei[e]; d = ei[E + e]; } else { s = d = e - E; }
    float x = asr[s * heads + h] + ads[d * heads + h];
    float v = (x >= 0.0f) ? x : 0.2f * x;   // leaky_relu 0.2
    ev[(size_t)e * heads + h] = v;
    atomicMaxF(&mbuf[d * heads + h], v);
}

__global__ void gat_edge_exp(const int* __restrict__ ei, int E,
                             const float* __restrict__ ev, const float* __restrict__ mbuf,
                             float* __restrict__ pbuf, float* __restrict__ sbuf,
                             int heads, int total)
{
    int idx = blockIdx.x * blockDim.x + threadIdx.x;
    if (idx >= total) return;
    int e = idx / heads, h = idx % heads;
    int d = (e < E) ? ei[E + e] : (e - E);
    float p = expf(ev[(size_t)e * heads + h] - mbuf[d * heads + h]);
    pbuf[(size_t)e * heads + h] = p;
    atomicAdd(&sbuf[d * heads + h], p);
}

__global__ __launch_bounds__(256) void gat_aggregate(
    const int* __restrict__ ei, int E,
    const float* __restrict__ hfeat, const float* __restrict__ pbuf, const float* __restrict__ sbuf,
    float* __restrict__ out, int heads, int ch)
{
    int e = blockIdx.x;
    int s, d;
    if (e < E) { s = ei[e]; d = ei[E + e]; } else { s = d = e - E; }
    for (int h = 0; h < heads; ++h) {
        float alpha = pbuf[(size_t)e * heads + h] / (sbuf[d * heads + h] + 1e-16f);
        for (int c = threadIdx.x; c < ch; c += blockDim.x)
            atomicAdd(&out[(size_t)d * heads * ch + h * ch + c],
                      alpha * hfeat[(size_t)s * heads * ch + h * ch + c]);
    }
}

__global__ void gat_finalize(const float* __restrict__ in, const float* __restrict__ b,
                             float* __restrict__ out, int dim, size_t n, int act)  // act: 0 none, 1 elu
{
    for (size_t i = (size_t)blockIdx.x * blockDim.x + threadIdx.x; i < n; i += (size_t)gridDim.x * blockDim.x) {
        float v = in[i] + b[i % dim];
        if (act == 1) v = (v > 0.0f) ? v : expm1f(v);
        out[i] = v;
    }
}

__global__ __launch_bounds__(128) void rownorm_kernel(const float* __restrict__ in, float* __restrict__ out, int D) {
    __shared__ float red[128];
    const int row = blockIdx.x;
    const int tid = threadIdx.x;
    float v = in[(size_t)row * D + tid];
    red[tid] = v * v; __syncthreads();
    for (int off = 64; off > 0; off >>= 1) {
        if (tid < off) red[tid] += red[tid + off];
        __syncthreads();
    }
    float nrm = fmaxf(sqrtf(red[0]), 1e-12f);
    out[(size_t)row * D + tid] = v / nrm;
}

// ---------------------------------------------------------------------------
// Host orchestration
// ---------------------------------------------------------------------------
extern "C" void kernel_launch(void* const* d_in, const int* in_sizes, int n_in,
                              void* d_out, int out_size, void* d_ws, size_t ws_size,
                              hipStream_t stream)
{
    (void)in_sizes; (void)n_in; (void)out_size; (void)ws_size;
    constexpr int N = 512, L = 128, D = 768, NH = 12, FF = 3072, E = 32768;
    constexpr int NT = N * L;                 // 65536 tokens
    constexpr int EE = E + N;                 // with self loops

    const int*   input_ids = (const int*)d_in[0];
    const int*   amask     = (const int*)d_in[1];
    const int*   eidx      = (const int*)d_in[2];
    const float* word_emb  = (const float*)d_in[3];
    const float* pos_emb   = (const float*)d_in[4];
    const float* type_emb  = (const float*)d_in[5];
    const float* ln_emb_s  = (const float*)d_in[6];
    const float* ln_emb_b  = (const float*)d_in[7];
    const float* Wqkv      = (const float*)d_in[8];
    const float* bqkv      = (const float*)d_in[9];
    const float* Wo        = (const float*)d_in[10];
    const float* bo        = (const float*)d_in[11];
    const float* ln1_s     = (const float*)d_in[12];
    const float* ln1_b     = (const float*)d_in[13];
    const float* Wff1      = (const float*)d_in[14];
    const float* bff1      = (const float*)d_in[15];
    const float* Wff2      = (const float*)d_in[16];
    const float* bff2      = (const float*)d_in[17];
    const float* ln2_s     = (const float*)d_in[18];
    const float* ln2_b     = (const float*)d_in[19];
    const float* pool_w    = (const float*)d_in[20];
    const float* pool_b    = (const float*)d_in[21];
    const float* gat1_w    = (const float*)d_in[22];
    const float* gat1_asrc = (const float*)d_in[23];
    const float* gat1_adst = (const float*)d_in[24];
    const float* gat1_b    = (const float*)d_in[25];
    const float* gat2_w    = (const float*)d_in[26];
    const float* gat2_asrc = (const float*)d_in[27];
    const float* gat2_adst = (const float*)d_in[28];
    const float* gat2_b    = (const float*)d_in[29];
    const float* proj_w1   = (const float*)d_in[30];
    const float* proj_b1   = (const float*)d_in[31];
    const float* proj_w2   = (const float*)d_in[32];
    const float* proj_b2   = (const float*)d_in[33];
    const float* cls_w     = (const float*)d_in[34];
    const float* cls_b     = (const float*)d_in[35];
    float* out = (float*)d_out;

    // ---- workspace carve (deterministic every call) ----
    char* ws = (char*)d_ws; size_t off = 0;
    auto alloc = [&](size_t bytes) -> char* {
        char* p = ws + off; off = (off + bytes + 255) & ~(size_t)255; return p;
    };
    unsigned short* Wqkv_b = (unsigned short*)alloc((size_t)12 * D * 3 * D * 2);
    unsigned short* Wo_b   = (unsigned short*)alloc((size_t)12 * D * D * 2);
    unsigned short* Wff1_b = (unsigned short*)alloc((size_t)12 * D * FF * 2);
    unsigned short* Wff2_b = (unsigned short*)alloc((size_t)12 * FF * D * 2);
    unsigned short* Wpool_b= (unsigned short*)alloc((size_t)D * D * 2);
    float*          tmpf   = (float*)alloc((size_t)NT * D * 4);
    float*          xf     = (float*)alloc((size_t)NT * D * 4);
    unsigned short* xb     = (unsigned short*)alloc((size_t)NT * D * 2);
    unsigned short* qkvb   = (unsigned short*)alloc((size_t)NT * 3 * D * 2);
    unsigned short* ctxb   = (unsigned short*)alloc((size_t)NT * D * 2);
    unsigned short* ffb    = (unsigned short*)alloc((size_t)NT * FF * 2);
    float* h_text = (float*)alloc((size_t)N * D * 4);
    float* h1g    = (float*)alloc((size_t)N * 1024 * 4);
    float* asr1   = (float*)alloc((size_t)N * 4 * 4);
    float* ads1   = (float*)alloc((size_t)N * 4 * 4);
    float* m1     = (float*)alloc((size_t)N * 4 * 4);
    float* s1     = (float*)alloc((size_t)N * 4 * 4);
    float* ev1    = (float*)alloc((size_t)EE * 4 * 4);
    float* p1     = (float*)alloc((size_t)EE * 4 * 4);
    float* out1   = (float*)alloc((size_t)N * 1024 * 4);
    float* h1     = (float*)alloc((size_t)N * 1024 * 4);
    float* h2g    = (float*)alloc((size_t)N * 128 * 4);
    float* asr2   = (float*)alloc((size_t)N * 4);
    float* ads2   = (float*)alloc((size_t)N * 4);
    float* m2     = (float*)alloc((size_t)N * 4);
    float* s2     = (float*)alloc((size_t)N * 4);
    float* ev2    = (float*)alloc((size_t)EE * 4);
    float* p2     = (float*)alloc((size_t)EE * 4);
    float* out2   = (float*)alloc((size_t)N * 128 * 4);
    float* h2     = (float*)alloc((size_t)N * 128 * 4);
    float* t1     = (float*)alloc((size_t)N * 128 * 4);
    float* fraw   = (float*)alloc((size_t)N * 128 * 4);

    const dim3 B256(256);
    // ---- convert weights to bf16 ----
    cvt_f32_bf16<<<2048, B256, 0, stream>>>(Wqkv, Wqkv_b, (size_t)12 * D * 3 * D);
    cvt_f32_bf16<<<2048, B256, 0, stream>>>(Wo,   Wo_b,   (size_t)12 * D * D);
    cvt_f32_bf16<<<2048, B256, 0, stream>>>(Wff1, Wff1_b, (size_t)12 * D * FF);
    cvt_f32_bf16<<<2048, B256, 0, stream>>>(Wff2, Wff2_b, (size_t)12 * FF * D);
    cvt_f32_bf16<<<1024, B256, 0, stream>>>(pool_w, Wpool_b, (size_t)D * D);

    // ---- embeddings + LN ----
    embed_kernel<<<NT, B256, 0, stream>>>(input_ids, word_emb, pos_emb, type_emb, tmpf);
    ln_kernel<<<NT, B256, 0, stream>>>(tmpf, ln_emb_s, ln_emb_b, xf, xb, D);

    // ---- 12 transformer layers ----
    for (int i = 0; i < 12; ++i) {
        // QKV: [NT,768]x[768,2304] -> bf16
        gemm_bf16_wmma<<<dim3(NT / 128, 18), B256, 0, stream>>>(
            xb, D, Wqkv_b + (size_t)i * D * 3 * D, 3 * D, bqkv + (size_t)i * 3 * D,
            nullptr, 0, nullptr, 0, qkvb, 3 * D, D, 0);
        // attention
        attn_kernel<<<dim3(NH, N), B256, 0, stream>>>(qkvb, amask, ctxb);
        // Wo + residual -> tmpf ; LN -> x
        gemm_bf16_wmma<<<dim3(NT / 128, 6), B256, 0, stream>>>(
            ctxb, D, Wo_b + (size_t)i * D * D, D, bo + (size_t)i * D,
            xf, D, tmpf, D, nullptr, 0, D, 0);
        ln_kernel<<<NT, B256, 0, stream>>>(tmpf, ln1_s + (size_t)i * D, ln1_b + (size_t)i * D, xf, xb, D);
        // FF1 + GELU -> bf16
        gemm_bf16_wmma<<<dim3(NT / 128, 24), B256, 0, stream>>>(
            xb, D, Wff1_b + (size_t)i * D * FF, FF, bff1 + (size_t)i * FF,
            nullptr, 0, nullptr, 0, ffb, FF, D, 1);
        // FF2 + residual -> tmpf ; LN -> x
        gemm_bf16_wmma<<<dim3(NT / 128, 6), B256, 0, stream>>>(
            ffb, FF, Wff2_b + (size_t)i * FF * D, D, bff2 + (size_t)i * D,
            xf, D, tmpf, D, nullptr, 0, FF, 0);
        ln_kernel<<<NT, B256, 0, stream>>>(tmpf, ln2_s + (size_t)i * D, ln2_b + (size_t)i * D, xf, xb, D);
    }

    // ---- pooler: tanh(x[:,0] @ pool_w + pool_b)  (A row stride = L*D) ----
    gemm_bf16_wmma<<<dim3(N / 128, 6), B256, 0, stream>>>(
        xb, L * D, Wpool_b, D, pool_b, nullptr, 0, h_text, D, nullptr, 0, D, 2);

    // ---- GAT layer 1 (heads=4, ch=256, concat) ----
    sgemm_naive<<<(N * 1024 + 255) / 256, B256, 0, stream>>>(
        h_text, D, gat1_w, 1024, nullptr, h1g, 1024, N, 1024, D, 0);
    gat_scores<<<N, B256, 0, stream>>>(h1g, gat1_asrc, gat1_adst, asr1, ads1, 4, 256);
    fill_f32<<<64, B256, 0, stream>>>(m1, -1e30f, (size_t)N * 4);
    fill_f32<<<64, B256, 0, stream>>>(s1, 0.0f, (size_t)N * 4);
    fill_f32<<<2048, B256, 0, stream>>>(out1, 0.0f, (size_t)N * 1024);
    gat_edge_logits<<<(EE * 4 + 255) / 256, B256, 0, stream>>>(eidx, E, asr1, ads1, ev1, m1, 4, EE * 4);
    gat_edge_exp<<<(EE * 4 + 255) / 256, B256, 0, stream>>>(eidx, E, ev1, m1, p1, s1, 4, EE * 4);
    gat_aggregate<<<EE, B256, 0, stream>>>(eidx, E, h1g, p1, s1, out1, 4, 256);
    gat_finalize<<<2048, B256, 0, stream>>>(out1, gat1_b, h1, 1024, (size_t)N * 1024, 1);  // ELU

    // ---- GAT layer 2 (heads=1, ch=128, mean==identity) ----
    sgemm_naive<<<(N * 128 + 255) / 256, B256, 0, stream>>>(
        h1, 1024, gat2_w, 128, nullptr, h2g, 128, N, 128, 1024, 0);
    gat_scores<<<N, B256, 0, stream>>>(h2g, gat2_asrc, gat2_adst, asr2, ads2, 1, 128);
    fill_f32<<<8, B256, 0, stream>>>(m2, -1e30f, (size_t)N);
    fill_f32<<<8, B256, 0, stream>>>(s2, 0.0f, (size_t)N);
    fill_f32<<<256, B256, 0, stream>>>(out2, 0.0f, (size_t)N * 128);
    gat_edge_logits<<<(EE + 255) / 256, B256, 0, stream>>>(eidx, E, asr2, ads2, ev2, m2, 1, EE);
    gat_edge_exp<<<(EE + 255) / 256, B256, 0, stream>>>(eidx, E, ev2, m2, p2, s2, 1, EE);
    gat_aggregate<<<EE, B256, 0, stream>>>(eidx, E, h2g, p2, s2, out2, 1, 128);
    gat_finalize<<<256, B256, 0, stream>>>(out2, gat2_b, h2, 128, (size_t)N * 128, 0);

    // ---- heads: logits, feat, h2 (output order: logits | feat | h2) ----
    sgemm_naive<<<(N * 2 + 255) / 256, B256, 0, stream>>>(
        h2, 128, cls_w, 2, cls_b, out, 2, N, 2, 128, 0);                      // logits
    sgemm_naive<<<(N * 128 + 255) / 256, B256, 0, stream>>>(
        h2, 128, proj_w1, 128, proj_b1, t1, 128, N, 128, 128, 1);             // relu
    sgemm_naive<<<(N * 128 + 255) / 256, B256, 0, stream>>>(
        t1, 128, proj_w2, 128, proj_b2, fraw, 128, N, 128, 128, 0);
    rownorm_kernel<<<N, dim3(128), 0, stream>>>(fraw, out + N * 2, 128);      // feat
    hipMemcpyAsync(out + N * 2 + N * 128, h2, (size_t)N * 128 * 4,
                   hipMemcpyDeviceToDevice, stream);                          // h2
}
